// Informer_41308995453033
// MI455X (gfx1250) — matrix operand order repaired and checked by reading, not compiled, not run on hardware
//
#include <hip/hip_runtime.h>
#include <hip/hip_bf16.h>

typedef __bf16 bf16_t;
typedef __attribute__((ext_vector_type(16))) __bf16 v16bf;
typedef __attribute__((ext_vector_type(8)))  __bf16 v8bf;
typedef __attribute__((ext_vector_type(8)))  float  v8f;
typedef int v4i __attribute__((vector_size(4 * sizeof(int))));

#define AS_GLOBAL __attribute__((address_space(1)))
#define AS_LDS    __attribute__((address_space(3)))

// CDNA5 async global->LDS staging (ASYNCcnt-tracked), with compile-safe probe.
#if defined(__has_builtin)
#if __has_builtin(__builtin_amdgcn_global_load_async_to_lds_b128) && \
    __has_builtin(__builtin_amdgcn_s_wait_asynccnt)
#define USE_ASYNC_LDS 1
#endif
#endif

namespace cfg {
constexpr int B   = 32;
constexpr int L   = 512;
constexpr int CIN = 144;
constexpr int D   = 512;
constexpr int H   = 8;
constexpr int DH  = 64;
constexpr int DFF = 2048;
constexpr int M   = B * L;     // 16384 token rows
constexpr int KEMB = 448;      // 3*144 = 432 padded to 448 (multiple of 32)
}

// =====================================================================
// Tiled bf16 WMMA GEMM:  C[M,N] = A[M,K](bf16) * Bt[N,K](bf16)^T + bias
// Epilogue: optional residual add, exact GELU, f32 and/or bf16 outputs.
// 256 threads = 8 wave32 waves; tile 128x128, K-step 32; LDS double buffer.
// Staging uses GLOBAL_LOAD_ASYNC_TO_LDS_B128 (ASYNCcnt) when available.
// =====================================================================
#define TM  128
#define TN  128
#define TKW 32

__device__ __forceinline__ v16bf frag_a(const bf16_t* tile, int r0, int lane) {
  // 16-bit A 16x32: lanes 0-15 -> row=lane, K in {0..7, 16..23};
  //                 lanes 16-31 -> same rows, K in {8..15, 24..31}.
  int row = r0 + (lane & 15);
  int kh  = (lane & 16) ? 8 : 0;
  const bf16_t* p = tile + row * TKW + kh;
  v8bf lo = *(const v8bf*)(p);
  v8bf hi = *(const v8bf*)(p + 16);
  return __builtin_shufflevector(lo, hi, 0,1,2,3,4,5,6,7,8,9,10,11,12,13,14,15);
}

__device__ __forceinline__ v16bf frag_b(const bf16_t* tile, int r0, int lane) {
  // 16-bit B 32x16: lanes 0-15 -> col=lane, K in {0..7, 16..23}; lanes 16-31 -> K in {8..15, 24..31}.
  // tile is stored N-major (row n holds K contiguous), so loads are contiguous.
  int row = r0 + (lane & 15);
  int kh  = (lane & 16) ? 8 : 0;
  const bf16_t* p = tile + row * TKW + kh;
  v8bf lo = *(const v8bf*)(p);
  v8bf hi = *(const v8bf*)(p + 16);
  return __builtin_shufflevector(lo, hi, 0,1,2,3,4,5,6,7,8,9,10,11,12,13,14,15);
}

__global__ __launch_bounds__(256) void gemm_bf16_kernel(
    const bf16_t* __restrict__ A, const bf16_t* __restrict__ Bt,
    const float* __restrict__ bias, const float* __restrict__ resid,
    float* __restrict__ Cf, bf16_t* __restrict__ Cb,
    int M, int N, int K, int gelu)
{
  __shared__ __align__(32) bf16_t sA[2][TM * TKW];
  __shared__ __align__(32) bf16_t sB[2][TN * TKW];

  const int tid  = threadIdx.x;
  const int lane = tid & 31;
  const int wave = tid >> 5;
  const int wm   = (wave >> 1) * 32;   // 0,32,64,96
  const int wn   = (wave & 1) * 64;    // 0,64

  const int m0 = blockIdx.y * TM;
  const int n0 = blockIdx.x * TN;

  // Each thread stages 16 bf16 (32B) per matrix per K-tile.
  const int srow = tid >> 1;           // 0..127
  const int scol = (tid & 1) * 16;     // 0 or 16

  v8f acc[2][4];
  const v8f vzero = {0.f,0.f,0.f,0.f,0.f,0.f,0.f,0.f};
  #pragma unroll
  for (int i = 0; i < 2; ++i)
    #pragma unroll
    for (int j = 0; j < 4; ++j) acc[i][j] = vzero;

  auto stage = [&](int buf, int k0) {
    const bf16_t* ga = A  + (size_t)(m0 + srow) * K + k0 + scol;
    const bf16_t* gb = Bt + (size_t)(n0 + srow) * K + k0 + scol;
    bf16_t* la = &sA[buf][srow * TKW + scol];
    bf16_t* lb = &sB[buf][srow * TKW + scol];
#if defined(USE_ASYNC_LDS)
    // Direct global->LDS DMA; wave keeps issuing WMMAs while data streams in.
    __builtin_amdgcn_global_load_async_to_lds_b128(
        (AS_GLOBAL v4i*)const_cast<bf16_t*>(ga),     (AS_LDS v4i*)la,       0, 0);
    __builtin_amdgcn_global_load_async_to_lds_b128(
        (AS_GLOBAL v4i*)const_cast<bf16_t*>(ga + 8), (AS_LDS v4i*)(la + 8), 0, 0);
    __builtin_amdgcn_global_load_async_to_lds_b128(
        (AS_GLOBAL v4i*)const_cast<bf16_t*>(gb),     (AS_LDS v4i*)lb,       0, 0);
    __builtin_amdgcn_global_load_async_to_lds_b128(
        (AS_GLOBAL v4i*)const_cast<bf16_t*>(gb + 8), (AS_LDS v4i*)(lb + 8), 0, 0);
#else
    const uint4* g4a = (const uint4*)ga;
    uint4* l4a = (uint4*)la;
    l4a[0] = g4a[0]; l4a[1] = g4a[1];
    const uint4* g4b = (const uint4*)gb;
    uint4* l4b = (uint4*)lb;
    l4b[0] = g4b[0]; l4b[1] = g4b[1];
#endif
  };

  auto stage_wait = [&]() {
#if defined(USE_ASYNC_LDS)
    __builtin_amdgcn_s_wait_asynccnt(0);
#endif
  };

  const int nk = K / TKW;
  stage(0, 0);
  stage_wait();
  __syncthreads();

  for (int kt = 0; kt < nk; ++kt) {
    const int buf = kt & 1;
    if (kt + 1 < nk) stage(buf ^ 1, (kt + 1) * TKW);
    if (kt + 2 < nk) {
      __builtin_prefetch(A  + (size_t)(m0 + srow) * K + (kt + 2) * TKW + scol, 0, 0);
      __builtin_prefetch(Bt + (size_t)(n0 + srow) * K + (kt + 2) * TKW + scol, 0, 0);
    }

    v16bf af[2], bfr[4];
    #pragma unroll
    for (int i = 0; i < 2; ++i) af[i] = frag_a(sA[buf], wm + 16 * i, lane);
    #pragma unroll
    for (int j = 0; j < 4; ++j) bfr[j] = frag_b(sB[buf], wn + 16 * j, lane);

    #pragma unroll
    for (int i = 0; i < 2; ++i)
      #pragma unroll
      for (int j = 0; j < 4; ++j)
        acc[i][j] = __builtin_amdgcn_wmma_f32_16x16x32_bf16(
            false, af[i], false, bfr[j], (short)0, acc[i][j], false, false);

    stage_wait();          // async tile for kt+1 landed in LDS
    __syncthreads();
  }

  // Epilogue: 32-bit C/D 16x16 layout: col = lane&15, VGPR r -> row r + 8*(lane>>4)
  const int cl = lane & 15;
  const int rh = (lane >> 4) * 8;
  #pragma unroll
  for (int i = 0; i < 2; ++i) {
    #pragma unroll
    for (int j = 0; j < 4; ++j) {
      const int col = n0 + wn + 16 * j + cl;
      const float bv = bias ? bias[col] : 0.f;
      #pragma unroll
      for (int r = 0; r < 8; ++r) {
        const int row = m0 + wm + 16 * i + rh + r;
        const size_t off = (size_t)row * N + col;
        float v = acc[i][j][r] + bv;
        if (resid) v += resid[off];
        if (gelu)  v = 0.5f * v * (1.f + erff(v * 0.70710678118654752f));
        if (Cf) Cf[off] = v;
        if (Cb) Cb[off] = (bf16_t)v;
      }
    }
  }
}

// =====================================================================
// Weight transpose + f32->bf16 convert (with K zero-pad):
//   src f32 [K,N] -> dst bf16 [N,Kpad]
// =====================================================================
__global__ void transpose_cvt_kernel(const float* __restrict__ src,
                                     bf16_t* __restrict__ dst,
                                     int K, int N, int Kpad)
{
  int gid = blockIdx.x * blockDim.x + threadIdx.x;
  if (gid >= N * Kpad) return;
  int n = gid / Kpad;
  int k = gid - n * Kpad;
  float v = (k < K) ? src[(size_t)k * N + n] : 0.f;
  dst[gid] = (bf16_t)v;
}

// Elementwise f32 -> bf16
__global__ void cvt_bf16_kernel(const float* __restrict__ src,
                                bf16_t* __restrict__ dst, int n)
{
  int gid = blockIdx.x * blockDim.x + threadIdx.x;
  if (gid < n) dst[gid] = (bf16_t)src[gid];
}

// =====================================================================
// Build packed embedding LHS: A_emb[b*L+l, 0:144]   = x_enc[b, (l-1)%L]
//                                         [144:288] = x_enc[b, l]
//                                         [288:432] = x_enc[b, (l+1)%L]
//                                         [432:448] = 0        (bf16)
// =====================================================================
__global__ void pack_emb_kernel(const float* __restrict__ x_enc,
                                bf16_t* __restrict__ Aemb)
{
  using namespace cfg;
  int gid = blockIdx.x * blockDim.x + threadIdx.x;
  if (gid >= M * KEMB) return;
  int row = gid / KEMB;
  int c   = gid - row * KEMB;
  int b = row / L, l = row - b * L;
  float v = 0.f;
  if (c < 432) {
    int g  = c / CIN;          // 0,1,2
    int ch = c - g * CIN;
    int lp = l + g - 1;
    lp = (lp < 0) ? L - 1 : (lp >= L ? 0 : lp);
    v = x_enc[((size_t)b * L + lp) * CIN + ch];
  }
  Aemb[gid] = (bf16_t)v;
}

// Add sinusoidal positional embedding; also emit bf16 copy of activations.
__global__ void posadd_kernel(float* __restrict__ x, bf16_t* __restrict__ xb)
{
  using namespace cfg;
  int gid = blockIdx.x * blockDim.x + threadIdx.x;
  if (gid >= M * D) return;
  int d = gid & (D - 1);
  int l = (gid >> 9) & (L - 1);
  int d2 = d & ~1;
  float freq = __expf(-(float)d2 * (9.210340371976184f / (float)D)); // ln(1e4)
  float ang  = (float)l * freq;
  float pe   = (d & 1) ? __cosf(ang) : __sinf(ang);
  float v = x[gid] + pe;
  x[gid]  = v;
  xb[gid] = (bf16_t)v;
}

// =====================================================================
// ProbSparse attention (FACTOR=1 -> U_part = u = 7)
// =====================================================================
__device__ __forceinline__ int samp_idx(int l, int j, int layer) {
  unsigned s = (unsigned)(l * 7 + j) * 2654435761u + (unsigned)layer * 0x9E3779B9u;
  s ^= s >> 16; s *= 0x85EBCA6Bu; s ^= s >> 13;
  return (int)(s & (unsigned)(cfg::L - 1));
}

// M[b,h,l] = max_j(q_l . k_samp_j) - sum_j(q_l . k_samp_j) / L
__global__ void attn_sample_kernel(const float* __restrict__ Q,
                                   const float* __restrict__ K,
                                   float* __restrict__ Msc, int layer)
{
  using namespace cfg;
  int gid = blockIdx.x * blockDim.x + threadIdx.x;   // over B*H*L
  if (gid >= B * H * L) return;
  int l = gid % L;
  int h = (gid / L) % H;
  int b = gid / (L * H);
  const float* q = Q + ((size_t)(b * L + l) * D) + h * DH;
  float mx = -1e30f, sm = 0.f;
  for (int j = 0; j < 7; ++j) {
    int ki = samp_idx(l, j, layer);
    const float* kv = K + ((size_t)(b * L + ki) * D) + h * DH;
    float dot = 0.f;
    #pragma unroll
    for (int e = 0; e < DH; ++e) dot += q[e] * kv[e];
    mx = fmaxf(mx, dot);
    sm += dot;
  }
  Msc[(size_t)(b * H + h) * L + l] = mx - sm / (float)L;
}

// One block per (b,h): top-7 queries, v-mean broadcast, softmax attention for
// the selected queries. ctx layout: [B*L, D] with column h*64+d (Wo-ready).
__global__ __launch_bounds__(256) void attn_top_kernel(
    const float* __restrict__ Q, const float* __restrict__ K,
    const float* __restrict__ V, const float* __restrict__ Msc,
    float* __restrict__ ctx)
{
  using namespace cfg;
  __shared__ float sM[L];
  __shared__ float red[256];
  __shared__ int   redi[256];
  __shared__ int   sel[7];
  __shared__ float svm[DH];
  __shared__ float sq[DH];
  __shared__ float sattn[L];
  __shared__ float ps[4][DH];

  const int bh  = blockIdx.x;
  const int h   = bh % H;
  const int b   = bh / H;
  const int tid = threadIdx.x;

  for (int i = tid; i < L; i += 256) sM[i] = Msc[(size_t)bh * L + i];
  __syncthreads();

  // --- top-7 argmax passes ---
  for (int p = 0; p < 7; ++p) {
    float best = -1e30f; int bi = 0;
    for (int i = tid; i < L; i += 256) {
      float v = sM[i];
      if (v > best) { best = v; bi = i; }
    }
    red[tid] = best; redi[tid] = bi;
    __syncthreads();
    for (int st = 128; st > 0; st >>= 1) {
      if (tid < st && red[tid + st] > red[tid]) {
        red[tid] = red[tid + st]; redi[tid] = redi[tid + st];
      }
      __syncthreads();
    }
    if (tid == 0) { sel[p] = redi[0]; sM[redi[0]] = -1e30f; }
    __syncthreads();
  }

  // --- v mean over L (per head) ---
  {
    int d = tid & 63, ch = tid >> 6;          // 4 chunks of 128 rows
    float s = 0.f;
    for (int l = ch * 128; l < (ch + 1) * 128; ++l)
      s += V[((size_t)(b * L + l) * D) + h * DH + d];
    ps[ch][d] = s;
    __syncthreads();
    if (tid < DH)
      svm[tid] = (ps[0][tid] + ps[1][tid] + ps[2][tid] + ps[3][tid]) * (1.f / (float)L);
    __syncthreads();
  }

  // --- broadcast mean context to all non-selected rows ---
  for (int i = tid; i < L * DH; i += 256) {
    int l = i >> 6, d = i & 63;
    bool skip = false;
    #pragma unroll
    for (int p = 0; p < 7; ++p) skip |= (l == sel[p]);
    if (!skip)
      ctx[((size_t)(b * L + l) * D) + h * DH + d] = svm[d];
  }
  __syncthreads();

  // --- full attention for the 7 selected queries ---
  for (int p = 0; p < 7; ++p) {
    const int row = sel[p];
    if (tid < DH) sq[tid] = Q[((size_t)(b * L + row) * D) + h * DH + tid];
    __syncthreads();

    for (int kk = tid; kk < L; kk += 256) {
      const float* kv = K + ((size_t)(b * L + kk) * D) + h * DH;
      float dot = 0.f;
      #pragma unroll
      for (int e = 0; e < DH; ++e) dot += sq[e] * kv[e];
      sattn[kk] = dot * 0.125f;                 // 1/sqrt(64)
    }
    __syncthreads();

    float mx = -1e30f;
    for (int i = tid; i < L; i += 256) mx = fmaxf(mx, sattn[i]);
    red[tid] = mx; __syncthreads();
    for (int st = 128; st > 0; st >>= 1) {
      if (tid < st) red[tid] = fmaxf(red[tid], red[tid + st]);
      __syncthreads();
    }
    mx = red[0]; __syncthreads();

    float se = 0.f;
    for (int i = tid; i < L; i += 256) {
      float e = __expf(sattn[i] - mx);
      sattn[i] = e;
      se += e;
    }
    red[tid] = se; __syncthreads();
    for (int st = 128; st > 0; st >>= 1) {
      if (tid < st) red[tid] += red[tid + st];
      __syncthreads();
    }
    const float inv = 1.f / red[0];
    __syncthreads();

    int d = tid & 63, ch = tid >> 6;
    float s = 0.f;
    for (int l = ch * 128; l < (ch + 1) * 128; ++l)
      s += sattn[l] * V[((size_t)(b * L + l) * D) + h * DH + d];
    ps[ch][d] = s;
    __syncthreads();
    if (tid < DH)
      ctx[((size_t)(b * L + row) * D) + h * DH + tid] =
          (ps[0][tid] + ps[1][tid] + ps[2][tid] + ps[3][tid]) * inv;
    __syncthreads();
  }
}

// =====================================================================
// LayerNorm over D=512 (one row per block). Optional f32 out, bf16 out,
// and per-row scale (final layer fuses the x_mark multiply).
// =====================================================================
__global__ __launch_bounds__(256) void layernorm_kernel(
    const float* __restrict__ in, const float* __restrict__ g,
    const float* __restrict__ be, float* __restrict__ outf,
    bf16_t* __restrict__ outb, const float* __restrict__ rowscale)
{
  using namespace cfg;
  __shared__ float red[256];
  const int row = blockIdx.x;
  const int tid = threadIdx.x;
  const float* x = in + (size_t)row * D;

  float s = 0.f;
  for (int c = tid; c < D; c += 256) s += x[c];
  red[tid] = s; __syncthreads();
  for (int st = 128; st > 0; st >>= 1) {
    if (tid < st) red[tid] += red[tid + st];
    __syncthreads();
  }
  const float mu = red[0] / (float)D;
  __syncthreads();

  float vs = 0.f;
  for (int c = tid; c < D; c += 256) { float d = x[c] - mu; vs += d * d; }
  red[tid] = vs; __syncthreads();
  for (int st = 128; st > 0; st >>= 1) {
    if (tid < st) red[tid] += red[tid + st];
    __syncthreads();
  }
  const float rstd = rsqrtf(red[0] / (float)D + 1e-5f);
  const float rs   = rowscale ? rowscale[row] : 1.f;

  for (int c = tid; c < D; c += 256) {
    float y = ((x[c] - mu) * rstd * g[c] + be[c]) * rs;
    size_t off = (size_t)row * D + c;
    if (outf) outf[off] = y;
    if (outb) outb[off] = (bf16_t)y;
  }
}

// =====================================================================
// Host orchestration
// =====================================================================
extern "C" void kernel_launch(void* const* d_in, const int* in_sizes, int n_in,
                              void* d_out, int out_size, void* d_ws, size_t ws_size,
                              hipStream_t stream)
{
  using namespace cfg;
  const float* x_enc  = (const float*)d_in[0];
  const float* x_mark = (const float*)d_in[1];
  const float* emb_w  = (const float*)d_in[2];   // (3,144,512) == (432,512)
  const float* Wq = (const float*)d_in[3];  const float* bq = (const float*)d_in[4];
  const float* Wk = (const float*)d_in[5];  const float* bk = (const float*)d_in[6];
  const float* Wv = (const float*)d_in[7];  const float* bv = (const float*)d_in[8];
  const float* Wo = (const float*)d_in[9];  const float* bo = (const float*)d_in[10];
  const float* W1 = (const float*)d_in[11]; const float* b1 = (const float*)d_in[12];
  const float* W2 = (const float*)d_in[13]; const float* b2 = (const float*)d_in[14];
  const float* ln1_g = (const float*)d_in[15]; const float* ln1_b = (const float*)d_in[16];
  const float* ln2_g = (const float*)d_in[17]; const float* ln2_b = (const float*)d_in[18];
  const float* lnf_g = (const float*)d_in[19]; const float* lnf_b = (const float*)d_in[20];
  float* out = (float*)d_out;

  char* ws = (char*)d_ws;
  size_t off = 0;
  auto wsa = [&](size_t bytes) -> void* {
    void* p = ws + off;
    off += (bytes + 255) & ~(size_t)255;
    return p;
  };

  const size_t actF = (size_t)M * D * sizeof(float);   // 33.5 MB
  const size_t actB = (size_t)M * D * sizeof(bf16_t);  // 16.8 MB

  float*  X    = (float*) wsa(actF);
  bf16_t* XB   = (bf16_t*)wsa(actB);
  float*  Qf   = (float*) wsa(actF);          // also reused as pre-LN residual sum
  float*  Kf   = (float*) wsa(actF);          // FFN hidden (bf16) overlays Kf+Vf
  float*  Vf   = (float*) wsa(actF);
  float*  CTX  = (float*) wsa(actF);          // embedding LHS (bf16) overlays CTX
  bf16_t* CTXB = (bf16_t*)wsa(actB);
  float*  Msc  = (float*) wsa((size_t)B * H * L * sizeof(float));
  bf16_t* WqT  = (bf16_t*)wsa((size_t)D * D * sizeof(bf16_t));
  bf16_t* WkT  = (bf16_t*)wsa((size_t)D * D * sizeof(bf16_t));
  bf16_t* WvT  = (bf16_t*)wsa((size_t)D * D * sizeof(bf16_t));
  bf16_t* WoT  = (bf16_t*)wsa((size_t)D * D * sizeof(bf16_t));
  bf16_t* W1T  = (bf16_t*)wsa((size_t)DFF * D * sizeof(bf16_t));
  bf16_t* W2T  = (bf16_t*)wsa((size_t)D * DFF * sizeof(bf16_t));
  bf16_t* WembT= (bf16_t*)wsa((size_t)D * KEMB * sizeof(bf16_t));
  (void)ws_size; (void)in_sizes; (void)n_in; (void)out_size;

  bf16_t* AEMB = (bf16_t*)CTX;   // overlay: only needed before attention
  bf16_t* HB   = (bf16_t*)Kf;    // overlay: FFN hidden (16384x2048 bf16) over Kf+Vf

  auto gemm = [&](const bf16_t* A, const bf16_t* Bt, const float* bias,
                  const float* resid, float* Cf, bf16_t* Cb,
                  int Mm, int Nn, int Kk, int gelu) {
    dim3 grid(Nn / TN, Mm / TM);
    gemm_bf16_kernel<<<grid, 256, 0, stream>>>(A, Bt, bias, resid, Cf, Cb,
                                               Mm, Nn, Kk, gelu);
  };

  // ---- embedding ----
  pack_emb_kernel<<<(M * KEMB + 255) / 256, 256, 0, stream>>>(x_enc, AEMB);
  transpose_cvt_kernel<<<(D * KEMB + 255) / 256, 256, 0, stream>>>(emb_w, WembT, 432, D, KEMB);
  gemm(AEMB, WembT, nullptr, nullptr, X, nullptr, M, D, KEMB, 0);
  posadd_kernel<<<(M * D + 255) / 256, 256, 0, stream>>>(X, XB);

  // ---- encoder layers ----
  for (int i = 0; i < 2; ++i) {
    const float* Wq_i = Wq + (size_t)i * D * D;   const float* bq_i = bq + (size_t)i * D;
    const float* Wk_i = Wk + (size_t)i * D * D;   const float* bk_i = bk + (size_t)i * D;
    const float* Wv_i = Wv + (size_t)i * D * D;   const float* bv_i = bv + (size_t)i * D;
    const float* Wo_i = Wo + (size_t)i * D * D;   const float* bo_i = bo + (size_t)i * D;
    const float* W1_i = W1 + (size_t)i * D * DFF; const float* b1_i = b1 + (size_t)i * DFF;
    const float* W2_i = W2 + (size_t)i * DFF * D; const float* b2_i = b2 + (size_t)i * D;

    transpose_cvt_kernel<<<(D * D + 255) / 256, 256, 0, stream>>>(Wq_i, WqT, D, D, D);
    transpose_cvt_kernel<<<(D * D + 255) / 256, 256, 0, stream>>>(Wk_i, WkT, D, D, D);
    transpose_cvt_kernel<<<(D * D + 255) / 256, 256, 0, stream>>>(Wv_i, WvT, D, D, D);
    transpose_cvt_kernel<<<(D * D + 255) / 256, 256, 0, stream>>>(Wo_i, WoT, D, D, D);
    transpose_cvt_kernel<<<(DFF * D + 255) / 256, 256, 0, stream>>>(W1_i, W1T, D, DFF, D);
    transpose_cvt_kernel<<<(D * DFF + 255) / 256, 256, 0, stream>>>(W2_i, W2T, DFF, D, DFF);

    // QKV projections
    gemm(XB, WqT, bq_i, nullptr, Qf, nullptr, M, D, D, 0);
    gemm(XB, WkT, bk_i, nullptr, Kf, nullptr, M, D, D, 0);
    gemm(XB, WvT, bv_i, nullptr, Vf, nullptr, M, D, D, 0);

    // ProbSparse attention
    attn_sample_kernel<<<(B * H * L + 255) / 256, 256, 0, stream>>>(Qf, Kf, Msc, i);
    attn_top_kernel<<<B * H, 256, 0, stream>>>(Qf, Kf, Vf, Msc, CTX);
    cvt_bf16_kernel<<<(M * D + 255) / 256, 256, 0, stream>>>(CTX, CTXB, M * D);

    // output projection + residual, then LN1 -> X / XB
    gemm(CTXB, WoT, bo_i, X, Qf, nullptr, M, D, D, 0);
    layernorm_kernel<<<M, 256, 0, stream>>>(Qf, ln1_g + (size_t)i * D, ln1_b + (size_t)i * D,
                                            X, XB, nullptr);

    // FFN: GELU(x@W1+b1)@W2+b2 + residual, then LN2 -> X / XB
    gemm(XB, W1T, b1_i, nullptr, nullptr, HB, M, DFF, D, 1);
    gemm(HB, W2T, b2_i, X, Qf, nullptr, M, D, DFF, 0);
    layernorm_kernel<<<M, 256, 0, stream>>>(Qf, ln2_g + (size_t)i * D, ln2_b + (size_t)i * D,
                                            X, XB, nullptr);
  }

  // ---- final LN fused with x_mark scaling, straight to d_out ----
  layernorm_kernel<<<M, 256, 0, stream>>>(X, lnf_g, lnf_b, out, nullptr, x_mark);
}
